// GNNLayer_77498389889587
// MI455X (gfx1250) — matrix operand order repaired and checked by reading, not compiled, not run on hardware
//
#include <hip/hip_runtime.h>
#include <hip/hip_bf16.h>
#include <stdint.h>

// ---------------------------------------------------------------------------
// CDNA5 (gfx1250) fused GNN layer: bf16 WMMA (v_wmma_f32_16x16x32_bf16) GEMMs
// with f32 accumulation, segment-softmax attention via L2 float atomics.
// ---------------------------------------------------------------------------

typedef __attribute__((ext_vector_type(16))) __bf16 v16bf;
typedef __attribute__((ext_vector_type(8)))  float  v8f;

union Frag16 {
    v16bf v;
    uint4 q[2];
    unsigned short h[16];
};

__device__ __forceinline__ unsigned short f32_to_bf16(float x) {
    unsigned u = __float_as_uint(x);
    unsigned r = u + 0x7FFFu + ((u >> 16) & 1u);   // round-to-nearest-even-ish
    return (unsigned short)(r >> 16);
}

__device__ __forceinline__ void lds_fence() {
    // CDNA5 split dep counter: wait for all outstanding DS ops (also a
    // compiler memory barrier so LDS producer->consumer order is kept).
    __asm volatile("s_wait_dscnt 0" ::: "memory");
}

__device__ __forceinline__ void zero_acc(v8f* acc, int n) {
#pragma unroll
    for (int i = 0; i < 8; ++i) {
        if (i >= n) break;
#pragma unroll
        for (int j = 0; j < 8; ++j) acc[i][j] = 0.0f;
    }
}

// One wave: D[16 x NT*16] += bf16 A[16 x KT*32] (LDS, row stride strideH halfs,
// multiple of 8) @ B (pre-packed WMMA fragments in global memory).
template <int KT, int NT>
__device__ __forceinline__ void wave_gemm(const unsigned short* a_lds, int strideH,
                                          const unsigned short* __restrict__ bpack,
                                          v8f* acc) {
    const int lane  = threadIdx.x & 31;
    const int row   = lane & 15;
    const int kbase = (lane < 16) ? 0 : 8;   // ISA 16-bit A layout
#pragma unroll
    for (int kt = 0; kt < KT; ++kt) {
        Frag16 a;
        const unsigned short* ap = a_lds + row * strideH + kt * 32 + kbase;
        a.q[0] = *reinterpret_cast<const uint4*>(ap);        // K = kbase..kbase+7
        a.q[1] = *reinterpret_cast<const uint4*>(ap + 16);   // K = kbase+16..+23
#pragma unroll
        for (int nt = 0; nt < NT; ++nt) {
            Frag16 b;
            const unsigned short* bp = bpack + ((size_t)(kt * NT + nt) * 32 + lane) * 16;
            b.q[0] = *reinterpret_cast<const uint4*>(bp);
            b.q[1] = *reinterpret_cast<const uint4*>(bp + 8);
            acc[nt] = __builtin_amdgcn_wmma_f32_16x16x32_bf16(
                false, a.v, false, b.v, (short)0, acc[nt], false, false);
        }
    }
}

// acc (NT tiles of 16x16 f32) + bias [-> relu] -> bf16 LDS (row stride strideH)
template <int NT>
__device__ __forceinline__ void store_acc_bf16(const v8f* acc, const float* __restrict__ bias,
                                               bool relu, unsigned short* out_lds, int strideH) {
    const int lane  = threadIdx.x & 31;
    const int col0  = lane & 15;
    const int rbase = (lane < 16) ? 0 : 8;   // ISA 32-bit C/D layout
#pragma unroll
    for (int nt = 0; nt < NT; ++nt) {
        const int col = nt * 16 + col0;
        const float b = bias[col];
#pragma unroll
        for (int v = 0; v < 8; ++v) {
            float x = acc[nt][v] + b;
            if (relu) x = fmaxf(x, 0.0f);
            out_lds[(rbase + v) * strideH + col] = f32_to_bf16(x);
        }
    }
}

__device__ __forceinline__ void atomicMaxF(float* addr, float val) {
    if (val >= 0.0f) atomicMax(reinterpret_cast<int*>(addr), __float_as_int(val));
    else             atomicMin(reinterpret_cast<unsigned int*>(addr), __float_as_uint(val));
}

// ---------------------------------------------------------------------------
// Weight packer: f32 [K x N] row-major -> bf16 WMMA-B fragments.
// grid.x = (K/32)*(N/16) fragments, 512 threads (= 32 lanes x 16 halfs).
// Lane L holds column nt*16+(L&15); half j holds K = kt*32 + 16*(j>=8) + 8*(L>=16) + (j&7).
// ---------------------------------------------------------------------------
__global__ void gnn_pack_w_kernel(const float* __restrict__ W, unsigned short* __restrict__ dst,
                                  int K, int N) {
    const int ntiles = N >> 4;
    const int frag   = blockIdx.x;
    const int kt     = frag / ntiles;
    const int nt     = frag - kt * ntiles;
    const int idx    = threadIdx.x;          // 0..511
    const int lane   = idx >> 4;
    const int j      = idx & 15;
    const int col    = nt * 16 + (lane & 15);
    const int k      = kt * 32 + ((j & 8) << 1) + ((lane >= 16) ? 8 : 0) + (j & 7);
    dst[(size_t)frag * 512 + idx] = f32_to_bf16(W[(size_t)k * N + col]);
}

__global__ void gnn_init_kernel(float* __restrict__ segmax, float* __restrict__ denom,
                                float* __restrict__ agg, int n_nodes) {
    const int i = blockIdx.x * blockDim.x + threadIdx.x;
    if (i < n_nodes * 128) agg[i] = 0.0f;
    if (i < n_nodes) {
        segmax[i] = __int_as_float(0xFF800000u);   // -inf
        denom[i]  = 0.0f;
    }
}

// ---------------------------------------------------------------------------
// Edge kernel: msg MLP (160->128 relu ->128 ->128) + attention gate, fused.
// One wave per 16-edge tile; 4 waves / 128-thread block.
// ---------------------------------------------------------------------------
#define EDGE_WAVES 4
__global__ __launch_bounds__(128) void gnn_edge_kernel(
    const float* __restrict__ nodes, const float* __restrict__ edges,
    const int* __restrict__ senders, const int* __restrict__ receivers,
    const unsigned short* __restrict__ wm0, const float* __restrict__ bm0,
    const unsigned short* __restrict__ wm1, const float* __restrict__ bm1,
    const unsigned short* __restrict__ wm2, const float* __restrict__ bm2,
    const unsigned short* __restrict__ wa0, const float* __restrict__ ba0,
    const float* __restrict__ wa1, const float* __restrict__ ba1,
    float* __restrict__ msgs, float* __restrict__ gates, float* __restrict__ segmax,
    int n_edges) {
    __shared__ __align__(16) unsigned short bufA[EDGE_WAVES][16 * 160];
    __shared__ __align__(16) unsigned short bufB[EDGE_WAVES][16 * 128];

    const int wave = threadIdx.x >> 5;
    const int lane = threadIdx.x & 31;
    const int tile = blockIdx.x * EDGE_WAVES + wave;
    const int e0   = tile * 16;
    if (e0 >= n_edges) return;
    const int eCount = min(16, n_edges - e0);

    unsigned short* A = bufA[wave];
    unsigned short* B = bufB[wave];

    // --- stage msg_input [16 x 160] as bf16: [edge(32) | send(64) | recv(64)]
    {
        const int row  = lane & 15;
        const int half = lane >> 4;                 // 2 lanes per row, 80 cols each
        const int e    = min(e0 + row, n_edges - 1);
        const int s    = senders[e];
        const int r    = receivers[e];
        const float* ef = edges + (size_t)e * 32;
        const float* sf = (s >= 0) ? nodes + (size_t)s * 64 : nullptr;
        const float* rf = (r >= 0) ? nodes + (size_t)r * 64 : nullptr;
        unsigned short* arow = A + row * 160;
        const float4 zz = make_float4(0.f, 0.f, 0.f, 0.f);
#pragma unroll
        for (int c4 = 0; c4 < 20; ++c4) {
            const int c = (half * 20 + c4) * 4;
            float4 x;
            if (c < 32)      x = *reinterpret_cast<const float4*>(ef + c);
            else if (c < 96) x = sf ? *reinterpret_cast<const float4*>(sf + c - 32) : zz;
            else             x = rf ? *reinterpret_cast<const float4*>(rf + c - 96) : zz;
            arow[c + 0] = f32_to_bf16(x.x);
            arow[c + 1] = f32_to_bf16(x.y);
            arow[c + 2] = f32_to_bf16(x.z);
            arow[c + 3] = f32_to_bf16(x.w);
        }
    }
    lds_fence();

    v8f acc[8];
    // --- layer 0: [16x160] @ [160x128], relu
    zero_acc(acc, 8);
    wave_gemm<5, 8>(A, 160, wm0, acc);
    store_acc_bf16<8>(acc, bm0, true, B, 128);
    lds_fence();
    // --- layer 1: [16x128] @ [128x128], no activation (matches reference)
    zero_acc(acc, 8);
    wave_gemm<4, 8>(B, 128, wm1, acc);
    store_acc_bf16<8>(acc, bm1, false, A, 128);  // reuse bufA with stride 128
    lds_fence();
    // --- layer 2: msgs = [16x128] @ [128x128] + bm2
    zero_acc(acc, 8);
    wave_gemm<4, 8>(A, 128, wm2, acc);
    {
        const int col0  = lane & 15;
        const int rbase = (lane < 16) ? 0 : 8;
#pragma unroll
        for (int nt = 0; nt < 8; ++nt) {
            const int col = nt * 16 + col0;
            const float b = bm2[col];
#pragma unroll
            for (int v = 0; v < 8; ++v) {
                const int row = rbase + v;
                const float x = acc[nt][v] + b;
                if (row < eCount) msgs[(size_t)(e0 + row) * 128 + col] = x;
                B[row * 128 + col] = f32_to_bf16(x);
            }
        }
    }
    lds_fence();

    // --- attention: g = msgs @ Wa0 + ba0 (16x64), gate = g @ Wa1 + ba1
    v8f acc2[4];
    zero_acc(acc2, 4);
    wave_gemm<4, 4>(B, 128, wa0, acc2);
    float* G = reinterpret_cast<float*>(A);       // 16x64 f32 scratch (4 KiB)
    {
        const int col0  = lane & 15;
        const int rbase = (lane < 16) ? 0 : 8;
#pragma unroll
        for (int nt = 0; nt < 4; ++nt) {
            const int col = nt * 16 + col0;
            const float b = ba0[col];
#pragma unroll
            for (int v = 0; v < 8; ++v) G[(rbase + v) * 64 + col] = acc2[nt][v] + b;
        }
    }
    lds_fence();
    if (lane < eCount) {
        float s = ba1[0];
        const float* grow = G + lane * 64;
#pragma unroll 8
        for (int c = 0; c < 64; ++c) s += grow[c] * wa1[c];
        const int e = e0 + lane;
        gates[e] = s;
        atomicMaxF(segmax + receivers[e], s);     // segment max
    }
}

__global__ void gnn_softmax_part_kernel(const float* __restrict__ gates,
                                        const int* __restrict__ receivers,
                                        const float* __restrict__ segmax,
                                        float* __restrict__ evals,
                                        float* __restrict__ denom, int n_edges) {
    const int e = blockIdx.x * blockDim.x + threadIdx.x;
    if (e >= n_edges) return;
    const int r = receivers[e];
    const float v = __expf(gates[e] - segmax[r]);
    evals[e] = v;
    atomicAdd(denom + r, v);
}

__global__ __launch_bounds__(256) void gnn_aggregate_kernel(
    const float* __restrict__ msgs, const float* __restrict__ evals,
    const float* __restrict__ denom, const int* __restrict__ receivers,
    float* __restrict__ agg, int n_edges) {
    const int t   = threadIdx.x;
    const int col = t & 127;
    const int e   = blockIdx.x * 2 + (t >> 7);
    if (e >= n_edges) return;
    const int r = receivers[e];
    const float attn = evals[e] / denom[r];
    atomicAdd(agg + (size_t)r * 128 + col, attn * msgs[(size_t)e * 128 + col]);
}

// ---------------------------------------------------------------------------
// Node kernel: update MLP on u = [nodes(64) | aggregated(128)] = 192 features.
// ---------------------------------------------------------------------------
#define NODE_WAVES 4
__global__ __launch_bounds__(128) void gnn_node_kernel(
    const float* __restrict__ nodes, const float* __restrict__ agg,
    const unsigned short* __restrict__ wu0, const float* __restrict__ bu0,
    const unsigned short* __restrict__ wu1, const float* __restrict__ bu1,
    const unsigned short* __restrict__ wu2, const float* __restrict__ bu2,
    float* __restrict__ out, int n_nodes) {
    __shared__ __align__(16) unsigned short bufA[NODE_WAVES][16 * 192];
    __shared__ __align__(16) unsigned short bufB[NODE_WAVES][16 * 128];

    const int wave = threadIdx.x >> 5;
    const int lane = threadIdx.x & 31;
    const int tile = blockIdx.x * NODE_WAVES + wave;
    const int n0   = tile * 16;
    if (n0 >= n_nodes) return;
    const int nCount = min(16, n_nodes - n0);

    unsigned short* A = bufA[wave];
    unsigned short* B = bufB[wave];

    {
        const int row  = lane & 15;
        const int half = lane >> 4;                 // 96 cols (24 float4) per lane
        const int n    = min(n0 + row, n_nodes - 1);
        const float* nf = nodes + (size_t)n * 64;
        const float* af = agg + (size_t)n * 128;
        unsigned short* arow = A + row * 192;
#pragma unroll
        for (int c4 = 0; c4 < 24; ++c4) {
            const int c = (half * 24 + c4) * 4;
            const float4 x = (c < 64) ? *reinterpret_cast<const float4*>(nf + c)
                                      : *reinterpret_cast<const float4*>(af + c - 64);
            arow[c + 0] = f32_to_bf16(x.x);
            arow[c + 1] = f32_to_bf16(x.y);
            arow[c + 2] = f32_to_bf16(x.z);
            arow[c + 3] = f32_to_bf16(x.w);
        }
    }
    lds_fence();

    v8f acc[8];
    zero_acc(acc, 8);
    wave_gemm<6, 8>(A, 192, wu0, acc);             // K = 192, relu
    store_acc_bf16<8>(acc, bu0, true, B, 128);
    lds_fence();
    zero_acc(acc, 8);
    wave_gemm<4, 8>(B, 128, wu1, acc);             // no activation
    store_acc_bf16<8>(acc, bu1, false, A, 128);
    lds_fence();
    zero_acc(acc, 8);
    wave_gemm<4, 8>(A, 128, wu2, acc);
    {
        const int col0  = lane & 15;
        const int rbase = (lane < 16) ? 0 : 8;
#pragma unroll
        for (int nt = 0; nt < 8; ++nt) {
            const int col = nt * 16 + col0;
            const float b = bu2[col];
#pragma unroll
            for (int v = 0; v < 8; ++v) {
                const int row = rbase + v;
                if (row < nCount) out[(size_t)(n0 + row) * 128 + col] = acc[nt][v] + b;
            }
        }
    }
}

// ---------------------------------------------------------------------------
extern "C" void kernel_launch(void* const* d_in, const int* in_sizes, int n_in,
                              void* d_out, int out_size, void* d_ws, size_t ws_size,
                              hipStream_t stream) {
    const float* nodes     = (const float*)d_in[0];
    const float* edges     = (const float*)d_in[1];
    const int*   senders   = (const int*)  d_in[2];
    const int*   receivers = (const int*)  d_in[3];
    const float* Wm0 = (const float*)d_in[4];  const float* bm0 = (const float*)d_in[5];
    const float* Wm1 = (const float*)d_in[6];  const float* bm1 = (const float*)d_in[7];
    const float* Wm2 = (const float*)d_in[8];  const float* bm2 = (const float*)d_in[9];
    const float* Wa0 = (const float*)d_in[10]; const float* ba0 = (const float*)d_in[11];
    const float* Wa1 = (const float*)d_in[12]; const float* ba1 = (const float*)d_in[13];
    const float* Wu0 = (const float*)d_in[14]; const float* bu0 = (const float*)d_in[15];
    const float* Wu1 = (const float*)d_in[16]; const float* bu1 = (const float*)d_in[17];
    const float* Wu2 = (const float*)d_in[18]; const float* bu2 = (const float*)d_in[19];

    const int n_nodes = in_sizes[0] / 64;
    const int n_edges = in_sizes[2];

    // ---- workspace carve-up (256B aligned) ----
    char* base = (char*)d_ws;
    size_t off = 0;
    auto carve = [&](size_t bytes) -> char* {
        off = (off + 255) & ~(size_t)255;
        char* p = base + off;
        off += bytes;
        return p;
    };
    unsigned short* pWm0 = (unsigned short*)carve((size_t)5 * 8 * 512 * 2);
    unsigned short* pWm1 = (unsigned short*)carve((size_t)4 * 8 * 512 * 2);
    unsigned short* pWm2 = (unsigned short*)carve((size_t)4 * 8 * 512 * 2);
    unsigned short* pWa0 = (unsigned short*)carve((size_t)4 * 4 * 512 * 2);
    unsigned short* pWu0 = (unsigned short*)carve((size_t)6 * 8 * 512 * 2);
    unsigned short* pWu1 = (unsigned short*)carve((size_t)4 * 8 * 512 * 2);
    unsigned short* pWu2 = (unsigned short*)carve((size_t)4 * 8 * 512 * 2);
    float* msgs   = (float*)carve((size_t)n_edges * 128 * 4);
    float* gates  = (float*)carve((size_t)n_edges * 4);
    float* evals  = (float*)carve((size_t)n_edges * 4);
    float* segmax = (float*)carve((size_t)n_nodes * 4);
    float* denom  = (float*)carve((size_t)n_nodes * 4);
    float* agg    = (float*)carve((size_t)n_nodes * 128 * 4);
    (void)ws_size; (void)n_in; (void)out_size;

    // ---- weight packing (bf16 WMMA-B fragments) ----
    gnn_pack_w_kernel<<<5 * 8, 512, 0, stream>>>(Wm0, pWm0, 160, 128);
    gnn_pack_w_kernel<<<4 * 8, 512, 0, stream>>>(Wm1, pWm1, 128, 128);
    gnn_pack_w_kernel<<<4 * 8, 512, 0, stream>>>(Wm2, pWm2, 128, 128);
    gnn_pack_w_kernel<<<4 * 4, 512, 0, stream>>>(Wa0, pWa0, 128, 64);
    gnn_pack_w_kernel<<<6 * 8, 512, 0, stream>>>(Wu0, pWu0, 192, 128);
    gnn_pack_w_kernel<<<4 * 8, 512, 0, stream>>>(Wu1, pWu1, 128, 128);
    gnn_pack_w_kernel<<<4 * 8, 512, 0, stream>>>(Wu2, pWu2, 128, 128);

    gnn_init_kernel<<<(n_nodes * 128 + 255) / 256, 256, 0, stream>>>(segmax, denom, agg, n_nodes);

    const int eTiles = (n_edges + 15) / 16;
    gnn_edge_kernel<<<(eTiles + EDGE_WAVES - 1) / EDGE_WAVES, 128, 0, stream>>>(
        nodes, edges, senders, receivers,
        pWm0, bm0, pWm1, bm1, pWm2, bm2, pWa0, ba0, Wa1, ba1,
        msgs, gates, segmax, n_edges);

    gnn_softmax_part_kernel<<<(n_edges + 255) / 256, 256, 0, stream>>>(
        gates, receivers, segmax, evals, denom, n_edges);

    gnn_aggregate_kernel<<<(n_edges + 1) / 2, 256, 0, stream>>>(
        msgs, evals, denom, receivers, agg, n_edges);

    const int nTiles = (n_nodes + 15) / 16;
    gnn_node_kernel<<<(nTiles + NODE_WAVES - 1) / NODE_WAVES, 128, 0, stream>>>(
        nodes, agg, pWu0, bu0, pWu1, bu1, pWu2, bu2,
        (float*)d_out, n_nodes);
}
